// DynamicDecoder_3453153706364
// MI455X (gfx1250) — matrix operand order repaired and checked by reading, not compiled, not run on hardware
//
#include <hip/hip_runtime.h>
#include <hip/hip_bf16.h>
#include <stdint.h>

// CDNA5 (gfx1250) wave32 WMMA types
typedef __attribute__((ext_vector_type(16))) __bf16 v16bf;
typedef __attribute__((ext_vector_type(8)))  float  v8f;

#define MROWS    64       // batch rows per workgroup (4 row tiles of 16)
#define ASTRIDE  520      // 512 + 8 bf16 pad -> activation row stride 16B-aligned

// Pre-transposed bf16 weight layouts in d_ws (n-major, k-contiguous):
//   W1t [512][512] @ 0
//   W2t [256][512] @ 262144
//   W3t [128][256] @ 393216
//   W4t [ 32][128] @ 425984  (rows 20..31 zero-padded)
#define W1T_OFF 0
#define W2T_OFF 262144
#define W3T_OFF 393216
#define W4T_OFF 425984

union Frag16 {
  v16bf v;
  uint4 u[2];
  __bf16 e[16];
};

// ---------------------------------------------------------------------------
// Prep: fp32 W[K][N] (row-major) -> bf16 Wt[Npad][K] (n-major), zero padding.
// Runs once per launch; ~860KB total, negligible vs main traffic.
// ---------------------------------------------------------------------------
__global__ void prep_wt_kernel(const float* __restrict__ W, __bf16* __restrict__ Wt,
                               int K, int N, int Npad) {
  const int idx = blockIdx.x * 256 + threadIdx.x;
  if (idx >= Npad * K) return;
  const int n = idx / K;
  const int k = idx - n * K;
  const float v = (n < N) ? W[(size_t)k * N + n] : 0.0f;
  Wt[idx] = (__bf16)v;
}

// ---------------------------------------------------------------------------
// One dense layer: OUT[64][N] = act( IN[64][K] @ W[K][N] + bias ).
// B fragments load straight from global bf16 Wt (L2-resident); A fragments
// from LDS. No barriers inside the k-loop; one __syncthreads per layer.
// ---------------------------------------------------------------------------
template<int K, int N, int CHUNK, bool LAST>
__device__ __forceinline__ void mlp_layer(
    const __bf16* inAct, __bf16* outAct, float* probsLds,
    const __bf16* __restrict__ Wt, const float* __restrict__ bias, int tid)
{
  constexpr int NT  = CHUNK / 16;   // column tiles per chunk
  constexpr int WPC = 8 / NT;       // waves stacked per column tile
  constexpr int RPW = 4 / WPC;      // row tiles per wave

  const int lane = tid & 31;
  const int wave = tid >> 5;
  const int half = lane >> 4;       // 16-lane half of the wave
  const int nIdx = lane & 15;
  const int col  = wave % NT;       // which column tile this wave owns
  const int rB   = (wave / NT) * RPW;
  const int colL = col * 16 + nIdx;

  for (int nc = 0; nc < (N + CHUNK - 1) / CHUNK; ++nc) {
    const int ncBase = nc * CHUNK;
    const __bf16* wRow = Wt + (size_t)(ncBase + colL) * K;

    v8f acc[RPW];
#pragma unroll
    for (int r = 0; r < RPW; ++r)
#pragma unroll
      for (int v = 0; v < 8; ++v) acc[r][v] = 0.0f;

#pragma unroll 2
    for (int ks = 0; ks < K / 32; ++ks) {
      // B fragment (32x16 bf16): lane = column, b[0..7]=W[k0+half*8..+7][n],
      // b[8..15]=W[k0+16+half*8..+7][n]
      // -> 2x global_load_b128 from the transposed row (64B line per 2 lanes)
      Frag16 bf;
      const __bf16* bp = wRow + ks * 32 + half * 8;
      bf.u[0] = *(const uint4*)(bp);
      bf.u[1] = *(const uint4*)(bp + 16);
#pragma unroll
      for (int r = 0; r < RPW; ++r) {
        // A fragment (16x32 bf16): lane = row, a[0..7]=K(half*8..+7),
        // a[8..15]=K(16+half*8..+7) -> 2x ds_load_b128
        Frag16 af;
        const __bf16* ap = inAct + ((rB + r) * 16 + nIdx) * ASTRIDE
                           + ks * 32 + half * 8;
        af.u[0] = *(const uint4*)(ap);
        af.u[1] = *(const uint4*)(ap + 16);
        acc[r] = __builtin_amdgcn_wmma_f32_16x16x32_bf16(
            false, af.v, false, bf.v, (short)0, acc[r], false, false);
      }
    }

    const int nG = ncBase + colL;
    const float bv = (nG < N) ? bias[nG] : 0.0f;
#pragma unroll
    for (int r = 0; r < RPW; ++r)
#pragma unroll
      for (int v = 0; v < 8; ++v) {
        // C/D layout: VGPR v, half h -> row m = v + 8*h ; col = lane%16
        const int m = (rB + r) * 16 + v + 8 * half;
        const float x = acc[r][v] + bv;
        if constexpr (LAST) {
          if (colL < 20)
            probsLds[m * 20 + colL] = 1.0f / (1.0f + __expf(-x));
        } else {
          outAct[m * ASTRIDE + nG] = (__bf16)fmaxf(x, 0.0f);
        }
      }
  }
  __syncthreads();  // layer boundary: OUT complete before it is read / reused
}

// ---------------------------------------------------------------------------
// Fused decoder: z -> 4 dense layers -> sigmoid/mask/adjacency outputs.
// ---------------------------------------------------------------------------
__global__ __launch_bounds__(256, 1)
void dyndec_fused_kernel(const float* __restrict__ z,
                         const float* __restrict__ b1,
                         const float* __restrict__ b2,
                         const float* __restrict__ b3,
                         const float* __restrict__ b4,
                         const float* __restrict__ adjG,
                         const __bf16* __restrict__ wt,   // d_ws bf16 weights
                         float* __restrict__ out, int B)
{
  __shared__ __bf16 actA[MROWS * ASTRIDE];   // 66,560 B
  __shared__ __bf16 actB[MROWS * ASTRIDE];   // 66,560 B
  __shared__ float  probsLds[MROWS * 20];    //  5,120 B
  __shared__ float  adjL[400];               //  1,600 B  (~137 KB total)

  const int tid = threadIdx.x;
  const int rowBase = blockIdx.x * MROWS;

  for (int i = tid; i < 400; i += 256) adjL[i] = adjG[i];

  // Stage z tile fp32 -> bf16 (64 rows x 512 cols), float4 global loads.
  for (int i = tid; i < MROWS * 128; i += 256) {
    const int row = i >> 7;
    const int c4  = (i & 127) * 4;
    const float4 zv = *(const float4*)(z + (size_t)(rowBase + row) * 512 + c4);
    union { __bf16 h[4]; uint2 u; } p;
    p.h[0] = (__bf16)zv.x; p.h[1] = (__bf16)zv.y;
    p.h[2] = (__bf16)zv.z; p.h[3] = (__bf16)zv.w;
    *(uint2*)(actA + row * ASTRIDE + c4) = p.u;
  }
  __syncthreads();

  mlp_layer<512, 512, 128, false>(actA, actB, nullptr, wt + W1T_OFF, b1, tid);
  mlp_layer<512, 256, 128, false>(actB, actA, nullptr, wt + W2T_OFF, b2, tid);
  mlp_layer<256, 128, 128, false>(actA, actB, nullptr, wt + W3T_OFF, b3, tid);
  mlp_layer<128,  20,  32, true >(actB, nullptr, probsLds, wt + W4T_OFF, b4, tid);

  // Outputs, flat in return order: adj [B*400], mask [B*20], probs [B*20], loss [1]
  const size_t maskBase = (size_t)B * 400;
  const size_t probBase = (size_t)B * 420;

  for (int e = tid; e < MROWS * 20; e += 256) {
    const int row = e / 20, col = e % 20;
    const float p = probsLds[row * 20 + col];
    const size_t g = (size_t)(rowBase + row) * 20 + col;
    out[maskBase + g] = (p > 0.5f) ? 1.0f : 0.0f;
    out[probBase + g] = p;
  }
  for (int e = tid; e < MROWS * 400; e += 256) {
    const int row = e / 400, idx = e % 400;
    const int i = idx / 20, j = idx % 20;
    const float mi = (probsLds[row * 20 + i] > 0.5f) ? 1.0f : 0.0f;
    const float mj = (probsLds[row * 20 + j] > 0.5f) ? 1.0f : 0.0f;
    out[(size_t)(rowBase + row) * 400 + idx] = mi * mj * adjL[idx];
  }
  if (blockIdx.x == 0 && tid == 0) out[(size_t)B * 440] = 0.0f;
}

// ---------------------------------------------------------------------------
extern "C" void kernel_launch(void* const* d_in, const int* in_sizes, int n_in,
                              void* d_out, int out_size, void* d_ws, size_t ws_size,
                              hipStream_t stream) {
  (void)n_in; (void)out_size; (void)ws_size;
  const float* z  = (const float*)d_in[0];
  const float* W1 = (const float*)d_in[1];
  const float* b1 = (const float*)d_in[2];
  const float* W2 = (const float*)d_in[3];
  const float* b2 = (const float*)d_in[4];
  const float* W3 = (const float*)d_in[5];
  const float* b3 = (const float*)d_in[6];
  const float* W4 = (const float*)d_in[7];
  const float* b4 = (const float*)d_in[8];
  const float* adj = (const float*)d_in[9];
  float* out  = (float*)d_out;
  __bf16* wt  = (__bf16*)d_ws;

  // One-time (per launch) weight transpose + bf16 conversion into d_ws.
  auto prep = [&](const float* W, __bf16* Wt, int K, int N, int Npad) {
    const int total = Npad * K;
    prep_wt_kernel<<<(total + 255) / 256, 256, 0, stream>>>(W, Wt, K, N, Npad);
  };
  prep(W1, wt + W1T_OFF, 512, 512, 512);
  prep(W2, wt + W2T_OFF, 512, 256, 256);
  prep(W3, wt + W3T_OFF, 256, 128, 128);
  prep(W4, wt + W4T_OFF, 128,  20,  32);

  const int B = in_sizes[0] / 512;        // 65536
  dim3 grid(B / MROWS), block(256);
  dyndec_fused_kernel<<<grid, block, 0, stream>>>(
      z, b1, b2, b3, b4, adj, wt, out, B);
}
